// TransformerBlock_88167088652346
// MI455X (gfx1250) — matrix operand order repaired and checked by reading, not compiled, not run on hardware
//
#include <hip/hip_runtime.h>
#include <math.h>

typedef unsigned short u16;
typedef unsigned int   u32;
typedef __attribute__((ext_vector_type(4)))  u32    u32x4;
typedef __attribute__((ext_vector_type(4)))  int    i32x4;
typedef __attribute__((ext_vector_type(8)))  __bf16 bf16x8;
typedef __attribute__((ext_vector_type(16))) __bf16 bf16x16;
typedef __attribute__((ext_vector_type(8)))  float  f32x8;

#define TSEQ 2048
#define CD   1024
#define NH   16
#define HS   64
#define MROWS 8192   // B*T

// ---- CDNA5 async global->LDS path (guarded; falls back to sync staging) ----
#if defined(__HIP_DEVICE_COMPILE__) && defined(__has_builtin)
#  if __has_builtin(__builtin_amdgcn_global_load_async_to_lds_b128) && \
      __has_builtin(__builtin_amdgcn_s_wait_asynccnt)
#    define HAVE_ASYNC 1
#  endif
#endif
#ifndef HAVE_ASYNC
#  define HAVE_ASYNC 0
#endif

#if HAVE_ASYNC
typedef __attribute__((address_space(1))) i32x4 gas_i32x4;   // global
typedef __attribute__((address_space(3))) i32x4 las_i32x4;   // LDS
#define GLD_ASYNC_B128(gp, lp)                                                \
    __builtin_amdgcn_global_load_async_to_lds_b128(                           \
        (gas_i32x4*)(gp), (las_i32x4*)(lp), 0, 0)
#define WAIT_ASYNC() __builtin_amdgcn_s_wait_asynccnt(0)
#else
#define WAIT_ASYNC() ((void)0)
#endif

// ---------- helpers ----------
__device__ __forceinline__ u16 f2bf(float f) {           // round-to-nearest-even
    u32 u = __float_as_uint(f);
    u += 0x7FFFu + ((u >> 16) & 1u);
    return (u16)(u >> 16);
}

__device__ __forceinline__ bf16x8 ld8(const u16* p) {    // 16-byte aligned load of 8 bf16
    union { u32x4 q; bf16x8 h; } u;
    u.q = *reinterpret_cast<const u32x4*>(p);
    return u.h;
}

__device__ __forceinline__ bf16x16 mk16(bf16x8 a, bf16x8 b) {
    return __builtin_shufflevector(a, b, 0,1,2,3,4,5,6,7,8,9,10,11,12,13,14,15);
}

__device__ __forceinline__ f32x8 wmma_bf16(bf16x16 a, bf16x16 b, f32x8 c) {
    // (neg_a, A, neg_b, B, c_mod, C, reuse_a, reuse_b)
    return __builtin_amdgcn_wmma_f32_16x16x32_bf16(false, a, false, b, (short)0, c, false, false);
}

__device__ __forceinline__ f32x8 zero8() {
    f32x8 z = {0.f,0.f,0.f,0.f,0.f,0.f,0.f,0.f};
    return z;
}

// ---------- packing kernels ----------
__global__ void pack_f32_to_bf16(const float* __restrict__ src, u16* __restrict__ dst, long n) {
    long i = (long)blockIdx.x * blockDim.x + threadIdx.x;
    if (i < n) dst[i] = f2bf(src[i]);
}

// dst (N x K bf16, transposed) = src (K x N f32)
__global__ void pack_transpose(const float* __restrict__ src, u16* __restrict__ dst, int K, int N) {
    long i = (long)blockIdx.x * blockDim.x + threadIdx.x;
    if (i >= (long)K * N) return;
    int k = (int)(i % K);
    int n = (int)(i / K);
    dst[i] = f2bf(src[(size_t)k * N + n]);
}

// Wq/Wk/Wv (H,C,hs) -> stacked transposed (3C x C) bf16; 0.125 score scale baked into Wq.
__global__ void pack_qkv(const float* __restrict__ Wq, const float* __restrict__ Wk,
                         const float* __restrict__ Wv, u16* __restrict__ dst) {
    long i = (long)blockIdx.x * blockDim.x + threadIdx.x;   // i = (m*C + n)*C + c
    if (i >= 3L * CD * CD) return;
    int c = (int)(i & (CD - 1));
    int n = (int)((i >> 10) & (CD - 1));
    int m = (int)(i >> 20);
    int h = n >> 6, d = n & 63;
    const float* src = (m == 0) ? Wq : (m == 1) ? Wk : Wv;
    float v = src[((size_t)h * CD + c) * HS + d];
    if (m == 0) v *= 0.125f;                                // hs^-0.5
    dst[i] = f2bf(v);
}

// ---------- generic bf16 WMMA GEMM: C = A(MxK) * BT(NxK)^T (+bias)(+relu) ----------
// Block tile 64x256, 8 waves (2x4), wave tile 32x64 (2x4 wmma). Double-buffered LDS,
// one barrier per k-step; next tile staged (async when available) during compute.
// FLAGS: 1=bias, 2=relu, 4=bf16 output
template <int FLAGS>
__global__ __launch_bounds__(256) void gemm_bf16_kernel(const u16* __restrict__ A,
                                                        const u16* __restrict__ BT,
                                                        const float* __restrict__ bias,
                                                        void* __restrict__ Cout,
                                                        int M, int N, int K) {
    __shared__ __align__(16) u16 As[2 * 64 * 40];
    __shared__ __align__(16) u16 Bs[2 * 256 * 40];

    const int tid = threadIdx.x;
    const int lane = tid & 31, wv = tid >> 5;
    const int lane16 = lane & 15, half = lane >> 4;
    const int wm = wv & 1, wn = wv >> 1;                  // 2x4 wave grid
    const int m0 = blockIdx.y * 64;
    const int n0 = blockIdx.x * 256;

    f32x8 acc[2][4];
#pragma unroll
    for (int i = 0; i < 2; ++i)
#pragma unroll
        for (int j = 0; j < 4; ++j) acc[i][j] = zero8();

    const int arow = tid >> 2, acb = (tid & 3) * 8;       // A tile: 64 rows x 32 halfs
    const u16* aG = A  + (size_t)(m0 + arow) * K + acb;
    const u16* bG = BT + (size_t)(n0 + tid) * K;          // B tile: 256 rows x 32 halfs

    auto issue = [&](int kk, int buf) {
        u16* Ab = As + buf * (64 * 40);
        u16* Bb = Bs + buf * (256 * 40);
#if HAVE_ASYNC
        GLD_ASYNC_B128(aG + kk, Ab + arow * 40 + acb);
#pragma unroll
        for (int j = 0; j < 4; ++j)
            GLD_ASYNC_B128(bG + kk + j * 8, Bb + tid * 40 + j * 8);
#else
        *(u32x4*)(Ab + arow * 40 + acb) = *(const u32x4*)(aG + kk);
#pragma unroll
        for (int j = 0; j < 4; ++j)
            *(u32x4*)(Bb + tid * 40 + j * 8) = *(const u32x4*)(bG + kk + j * 8);
#endif
    };

    const int nk = K >> 5;
    issue(0, 0);
    int buf = 0;
    for (int t = 0; t < nk; ++t) {
        WAIT_ASYNC();                 // async tile writes visible (no-op on fallback)
        __syncthreads();              // tile[buf] ready; tile[buf^1] free to overwrite
        if (t + 1 < nk) issue((t + 1) << 5, buf ^ 1);

        const u16* Ab = As + buf * (64 * 40);
        const u16* Bb = Bs + buf * (256 * 40);
        bf16x16 aA[2], bB[4];
#pragma unroll
        for (int i = 0; i < 2; ++i) {                     // A: row=L%16, k = half*16 + e
            const u16* p = Ab + (wm * 32 + i * 16 + lane16) * 40 + half * 16;
            aA[i] = mk16(ld8(p), ld8(p + 8));
        }
#pragma unroll
        for (int j = 0; j < 4; ++j) {                     // B: col=L%16, k = half*16 + e
            const u16* p = Bb + (wn * 64 + j * 16 + lane16) * 40 + half * 16;
            bB[j] = mk16(ld8(p), ld8(p + 8));
        }
#pragma unroll
        for (int i = 0; i < 2; ++i)
#pragma unroll
            for (int j = 0; j < 4; ++j)
                acc[i][j] = wmma_bf16(aA[i], bB[j], acc[i][j]);
        buf ^= 1;
    }

#pragma unroll
    for (int i = 0; i < 2; ++i) {
#pragma unroll
        for (int j = 0; j < 4; ++j) {
            int n = n0 + wn * 64 + j * 16 + lane16;
            float bv = (FLAGS & 1) ? bias[n] : 0.0f;
#pragma unroll
            for (int r = 0; r < 8; ++r) {                 // D row = r + 8*half
                int m = m0 + wm * 32 + i * 16 + r + 8 * half;
                float v = acc[i][j][r] + bv;
                if (FLAGS & 2) v = fmaxf(v, 0.0f);
                if (FLAGS & 4) ((u16*)Cout)[(size_t)m * N + n] = f2bf(v);
                else           ((float*)Cout)[(size_t)m * N + n] = v;
            }
        }
    }
}

// ---------- causal flash attention (bf16 wmma, f32 online softmax) ----------
// qkv: (M x 3C) bf16, cols [0,C)=q (prescaled), [C,2C)=k, [2C,3C)=v. o_mat: (M x C) bf16.
__global__ __launch_bounds__(128) void flash_attn_kernel(const u16* __restrict__ qkv,
                                                         u16* __restrict__ o_mat) {
    __shared__ __align__(16) u16 Ks[32 * 72];      // K tile, row=s(32), col=d(64), pad->72
    __shared__ __align__(16) u16 Vt[64 * 40];      // V^T tile, row=d(64), col=s(32), pad->40
    __shared__ __align__(16) u16 Pw[4 * 16 * 40];  // per-wave P scratch 16x32, pad->40

    const int tid = threadIdx.x;
    const int lane = tid & 31, w = tid >> 5;
    const int lane16 = lane & 15, half = lane >> 4;

    const int qb = blockIdx.x;
    const int b = blockIdx.y / NH, h = blockIdx.y % NH;
    const int tBase = qb * 64 + w * 16;
    const int tMax  = tBase + 15;

    // Q in registers (A-operand layout): row = L%16
    const u16* qrow = qkv + (size_t)(b * TSEQ + tBase + lane16) * (3 * CD) + h * HS;
    bf16x16 aQ[2];
#pragma unroll
    for (int ka = 0; ka < 2; ++ka)
        aQ[ka] = mk16(ld8(qrow + ka * 32 + half * 16), ld8(qrow + ka * 32 + half * 16 + 8));

    f32x8 accO[4];
#pragma unroll
    for (int j = 0; j < 4; ++j) accO[j] = zero8();
    float rowMax[8], rowSum[8];
#pragma unroll
    for (int r = 0; r < 8; ++r) { rowMax[r] = -INFINITY; rowSum[r] = 0.0f; }

    const int kbLast = 2 * qb + 1;
    for (int kb = 0; kb <= kbLast; ++kb) {
        const int s0 = kb * 32;
        __syncthreads();
        {   // K tile (row-major s,d)
            int s = tid >> 2, dB = (tid & 3) * 16;
            const u16* kg = qkv + (size_t)(b * TSEQ + s0 + s) * (3 * CD) + CD + h * HS + dB;
#if HAVE_ASYNC
            GLD_ASYNC_B128(kg,     Ks + s * 72 + dB);
            GLD_ASYNC_B128(kg + 8, Ks + s * 72 + dB + 8);
#else
            *(u32x4*)(Ks + s * 72 + dB)     = *(const u32x4*)kg;
            *(u32x4*)(Ks + s * 72 + dB + 8) = *(const u32x4*)(kg + 8);
#endif
        }
        {   // V^T tile (d,s)
            int s = tid & 31, dB = (tid >> 5) * 16;
            const u16* vg = qkv + (size_t)(b * TSEQ + s0 + s) * (3 * CD) + 2 * CD + h * HS + dB;
#pragma unroll
            for (int jj = 0; jj < 16; ++jj) Vt[(dB + jj) * 40 + s] = vg[jj];
        }
        WAIT_ASYNC();
        __syncthreads();

        if (s0 <= tMax) {            // wave-uniform -> EXEC all ones for WMMA
            float sv[2][8];
#pragma unroll
            for (int j = 0; j < 2; ++j) {        // S subtile (16 rows x 16 cols)
                const u16* kr = Ks + (j * 16 + lane16) * 72 + half * 16;
                f32x8 accS = zero8();
                accS = wmma_bf16(aQ[0], mk16(ld8(kr),      ld8(kr + 8)),  accS);
                accS = wmma_bf16(aQ[1], mk16(ld8(kr + 32), ld8(kr + 40)), accS);
                int sCol = s0 + j * 16 + lane16;
#pragma unroll
                for (int r = 0; r < 8; ++r) {
                    int t = tBase + r + 8 * half;
                    sv[j][r] = (sCol <= t) ? accS[r] : -INFINITY;
                }
            }
            float scl[8];
#pragma unroll
            for (int r = 0; r < 8; ++r) {        // row max over 16-lane half-group
                float mx = fmaxf(sv[0][r], sv[1][r]);
#pragma unroll
                for (int off = 8; off; off >>= 1) mx = fmaxf(mx, __shfl_xor(mx, off, 32));
                float nm = fmaxf(rowMax[r], mx);
                scl[r] = __expf(rowMax[r] - nm);
                rowMax[r] = nm;
            }
            u16* pw = Pw + w * (16 * 40);
#pragma unroll
            for (int j = 0; j < 2; ++j)
#pragma unroll
                for (int r = 0; r < 8; ++r) {
                    float p = __expf(sv[j][r] - rowMax[r]);
                    sv[j][r] = p;
                    pw[(r + 8 * half) * 40 + j * 16 + lane16] = f2bf(p);
                }
#pragma unroll
            for (int r = 0; r < 8; ++r) {
                float ps = sv[0][r] + sv[1][r];
#pragma unroll
                for (int off = 8; off; off >>= 1) ps += __shfl_xor(ps, off, 32);
                rowSum[r] = rowSum[r] * scl[r] + ps;
            }
#pragma unroll
            for (int j2 = 0; j2 < 4; ++j2)
#pragma unroll
                for (int r = 0; r < 8; ++r) accO[j2][r] *= scl[r];

            // reload P in A-operand layout (same-wave LDS ops are in-order)
            const u16* pr = pw + lane16 * 40;
            bf16x16 aP = mk16(ld8(pr + half * 16), ld8(pr + half * 16 + 8));
#pragma unroll
            for (int j2 = 0; j2 < 4; ++j2) {     // O += P @ V
                const u16* vr = Vt + (j2 * 16 + lane16) * 40 + half * 16;
                accO[j2] = wmma_bf16(aP, mk16(ld8(vr), ld8(vr + 8)), accO[j2]);
            }
        }
    }

#pragma unroll
    for (int j2 = 0; j2 < 4; ++j2)
#pragma unroll
        for (int r = 0; r < 8; ++r) {
            int t = tBase + r + 8 * half;
            float o = accO[j2][r] / rowSum[r];
            o_mat[(size_t)(b * TSEQ + t) * CD + h * HS + j2 * 16 + lane16] = f2bf(o);
        }
}

// ---------- residual + custom LayerNorm (unbiased var, eps=1e-5) ----------
__global__ __launch_bounds__(256) void ln_residual_kernel(const float* __restrict__ xres,
                                                          const float* __restrict__ y,
                                                          float* __restrict__ outf,
                                                          u16* __restrict__ outb) {
    __shared__ float red[512];
    const int row = blockIdx.x, tid = threadIdx.x;
    const size_t base = (size_t)row * CD;
    float v[4], s = 0.f, ss = 0.f;
#pragma unroll
    for (int i = 0; i < 4; ++i) {
        v[i] = y[base + tid + i * 256];
        s += v[i]; ss += v[i] * v[i];
    }
    red[tid] = s; red[256 + tid] = ss;
    __syncthreads();
    for (int st = 128; st > 0; st >>= 1) {
        if (tid < st) { red[tid] += red[tid + st]; red[256 + tid] += red[256 + tid + st]; }
        __syncthreads();
    }
    float mean = red[0] * (1.0f / CD);
    float var  = (red[256] - CD * mean * mean) * (1.0f / (CD - 1));
    float inv  = rsqrtf(var + 1e-5f);
#pragma unroll
    for (int i = 0; i < 4; ++i) {
        int c = tid + i * 256;
        float o = xres[base + c] + (v[i] - mean) * inv;
        outf[base + c] = o;
        if (outb) outb[base + c] = f2bf(o);
    }
}

// ---------- launch ----------
extern "C" void kernel_launch(void* const* d_in, const int* in_sizes, int n_in,
                              void* d_out, int out_size, void* d_ws, size_t ws_size,
                              hipStream_t stream) {
    (void)in_sizes; (void)n_in; (void)out_size; (void)ws_size;
    const float* x  = (const float*)d_in[0];
    const float* Wq = (const float*)d_in[1];
    const float* Wk = (const float*)d_in[2];
    const float* Wv = (const float*)d_in[3];
    const float* Wo = (const float*)d_in[4];
    const float* bo = (const float*)d_in[5];
    const float* W1 = (const float*)d_in[6];
    const float* b1 = (const float*)d_in[7];
    const float* W2 = (const float*)d_in[8];
    const float* b2 = (const float*)d_in[9];

    char* ws = (char*)d_ws;                       // byte offsets (aliasing noted)
    u16*  WqkvT = (u16*)(ws + 0);                 //  6 MB  (3C x C)
    u16*  WoT   = (u16*)(ws + 6291456);           //  2 MB  (C x C)
    u16*  W1T   = (u16*)(ws + 8388608);           //  8 MB  (4C x C)
    u16*  W2T   = (u16*)(ws + 16777216);          //  8 MB  (C x 4C)
    u16*  xb    = (u16*)(ws + 25165824);          // 16 MB  (M x C)
    u16*  qkv   = (u16*)(ws + 41943040);          // 48 MB  (M x 3C)
    u16*  o_mat = (u16*)(ws + 92274688);          // 16 MB  (M x C)
    u16*  h1b   = (u16*)(ws + 41943040);          // 64 MB, aliases qkv+o_mat (both dead)
    float* attn = (float*)(ws + 109051904);       // 32 MB  (M x C)
    float* ff   = (float*)(ws + 109051904);       // 32 MB, aliases attn (dead)
    float* x1f  = (float*)(ws + 142606336);       // 32 MB
    u16*  x1b   = (u16*)(ws + 176160768);         // 16 MB  -> total ~184 MB

    // 1. pack activations + weights to bf16 (WMMA-friendly transposed layouts)
    pack_f32_to_bf16<<<32768, 256, 0, stream>>>(x, xb, (long)MROWS * CD);
    pack_qkv<<<12288, 256, 0, stream>>>(Wq, Wk, Wv, WqkvT);
    pack_transpose<<<4096,  256, 0, stream>>>(Wo, WoT, CD, CD);
    pack_transpose<<<16384, 256, 0, stream>>>(W1, W1T, CD, 4 * CD);
    pack_transpose<<<16384, 256, 0, stream>>>(W2, W2T, 4 * CD, CD);

    // 2. fused QKV projection: (M x 3C) bf16
    gemm_bf16_kernel<4><<<dim3(12, 128), 256, 0, stream>>>(xb, WqkvT, nullptr, qkv,
                                                           MROWS, 3 * CD, CD);
    // 3. causal flash attention -> concat-head o (M x C) bf16
    flash_attn_kernel<<<dim3(TSEQ / 64, 4 * NH), 128, 0, stream>>>(qkv, o_mat);

    // 4. output projection + bias -> f32
    gemm_bf16_kernel<1><<<dim3(4, 128), 256, 0, stream>>>(o_mat, WoT, bo, attn,
                                                          MROWS, CD, CD);
    // 5. x1 = x + LN(attn)
    ln_residual_kernel<<<MROWS, 256, 0, stream>>>(x, attn, x1f, x1b);

    // 6. h = relu(x1 @ W1 + b1) -> bf16
    gemm_bf16_kernel<7><<<dim3(16, 128), 256, 0, stream>>>(x1b, W1T, b1, h1b,
                                                           MROWS, 4 * CD, CD);
    // 7. ff = h @ W2 + b2 -> f32
    gemm_bf16_kernel<1><<<dim3(4, 128), 256, 0, stream>>>(h1b, W2T, b2, ff,
                                                          MROWS, CD, 4 * CD);
    // 8. out = x1 + LN(ff)
    ln_residual_kernel<<<MROWS, 256, 0, stream>>>(x1f, ff, (float*)d_out, nullptr);
}
